// AKTTransformerLayer_88381837017141
// MI455X (gfx1250) — matrix operand order repaired and checked by reading, not compiled, not run on hardware
//
#include <hip/hip_runtime.h>

// ---------------------------------------------------------------------------
// AKT transformer layer forward for MI455X (gfx1250, wave32, WMMA).
// All GEMMs use v_wmma_f32_16x16x32_bf16 (bf16 in, f32 accumulate).
// GEMM: 32x64 wave tile, double-buffered fragments (software pipelining),
// global_prefetch_b8 for the streaming A panels.
// Attention (incl. distance-decay cumsum) fused per 16-query-row tile with
// the score strip resident in LDS (~132KB of the 320KB WGP LDS).
// ---------------------------------------------------------------------------

#define NEGF (-1e32f)

typedef __attribute__((ext_vector_type(16))) __bf16 v16bf;
typedef __attribute__((ext_vector_type(8)))  float  v8f;

union BF16x16 {
    v16bf v;
    unsigned short u[16];
    uint4 q[2];
};

__device__ __forceinline__ unsigned short f32_to_bf16(float f) {
    unsigned int x = __float_as_uint(f);
    // round-to-nearest-even
    unsigned int r = (x + 0x7fffu + ((x >> 16) & 1u)) >> 16;
    return (unsigned short)r;
}

__device__ __forceinline__ float wave_sum(float v) {
    #pragma unroll
    for (int o = 16; o > 0; o >>= 1) v += __shfl_xor(v, o, 32);
    return v;
}
__device__ __forceinline__ float wave_max(float v) {
    #pragma unroll
    for (int o = 16; o > 0; o >>= 1) v = fmaxf(v, __shfl_xor(v, o, 32));
    return v;
}

// ---------------------------------------------------------------------------
// f32 -> bf16 conversion
// ---------------------------------------------------------------------------
__global__ void cvt_f32_bf16(const float* __restrict__ in,
                             unsigned short* __restrict__ out, size_t n) {
    size_t i = (size_t)blockIdx.x * blockDim.x + threadIdx.x;
    if (i < n) out[i] = f32_to_bf16(in[i]);
}

// ---------------------------------------------------------------------------
// V transpose: Vp (B*S, D) bf16 -> vT[b][h][d][s] bf16 (so attn·V B-fragments
// are contiguous along s).
// ---------------------------------------------------------------------------
__global__ void transpose_v(const unsigned short* __restrict__ Vp,
                            unsigned short* __restrict__ vT,
                            int Bb, int H, int S, int D) {
    size_t total = (size_t)Bb * D * S;
    size_t idx = (size_t)blockIdx.x * blockDim.x + threadIdx.x;
    if (idx >= total) return;
    int s = (int)(idx % S);
    size_t t = idx / S;
    int d = (int)(t % 64);
    size_t t2 = t / 64;
    int h = (int)(t2 % H);
    int b = (int)(t2 / H);
    vT[idx] = Vp[((size_t)b * S + s) * D + (size_t)h * 64 + d];
}

// ---------------------------------------------------------------------------
// Generic WMMA GEMM:  C[M,N] = A[M,K] * B[N,K]^T + bias  (x @ W.T form)
// A, Bm row-major bf16 with leading dim K.  Optional ReLU, optional f32 and/or
// bf16 output.
// Block = 8 waves; wave tile 32(M) x 64(N); block tile 64 x 256.
// Double-buffered fragments: loads for K-step k+32 issue before the 8 WMMAs
// of step k.  Streaming A panels are prefetched 8 K-steps ahead
// (global_prefetch_b8; B panels are WGP$-resident after the first M-block).
// Fragment layouts per CDNA5 ISA 7.12.2 (wave32):
//   A 16x32: lane(h=lane>>4, m=lane&15): elems 0..7=K(8h..), 8..15=K(16+8h..)
//   B 32x16: lane(h, n): elems 0..7=K(8h..), 8..15=K(16+8h..) of column n
//   C 16x16: elem r -> row r+8h, col n
// ---------------------------------------------------------------------------
__global__ void __launch_bounds__(256)
gemm_bf16_wmma(const unsigned short* __restrict__ A,
               const unsigned short* __restrict__ Bm,
               const float* __restrict__ bias,
               float* __restrict__ Cf,
               unsigned short* __restrict__ Cb,
               int M, int N, int K, int relu) {
    const int lane = threadIdx.x & 31;
    const int wv   = threadIdx.x >> 5;     // 0..7
    const int wm   = wv & 1;               // 2 waves along M
    const int wn   = wv >> 1;              // 4 waves along N
    const int m0   = blockIdx.y * 64 + wm * 32;
    const int n0   = blockIdx.x * 256 + wn * 64;
    const int h    = lane >> 4;
    const int l16  = lane & 15;

    v8f acc0[4] = {};   // rows m0..m0+15
    v8f acc1[4] = {};   // rows m0+16..m0+31

    const unsigned short* arow0 = A + (size_t)(m0 + l16) * K + 8 * h;
    const unsigned short* arow1 = A + (size_t)(m0 + 16 + l16) * K + 8 * h;
    const unsigned short* brow  = Bm + (size_t)(n0 + l16) * K + 8 * h;

    auto loadA = [&](BF16x16* a, int kb) {
        a[0].q[0] = *(const uint4*)(arow0 + kb);
        a[0].q[1] = *(const uint4*)(arow0 + kb + 16);
        a[1].q[0] = *(const uint4*)(arow1 + kb);
        a[1].q[1] = *(const uint4*)(arow1 + kb + 16);
    };
    auto loadB = [&](BF16x16* bb, int kb) {
        #pragma unroll
        for (int nt = 0; nt < 4; ++nt) {
            const unsigned short* bp = brow + (size_t)nt * 16 * K + kb;
            bb[nt].q[0] = *(const uint4*)(bp);
            bb[nt].q[1] = *(const uint4*)(bp + 16);
        }
    };
    auto mm = [&](BF16x16* a, BF16x16* bb) {
        #pragma unroll
        for (int nt = 0; nt < 4; ++nt) {
            acc0[nt] = __builtin_amdgcn_wmma_f32_16x16x32_bf16(
                false, a[0].v, false, bb[nt].v, (short)0, acc0[nt], false, false);
            acc1[nt] = __builtin_amdgcn_wmma_f32_16x16x32_bf16(
                false, a[1].v, false, bb[nt].v, (short)0, acc1[nt], false, false);
        }
    };

    BF16x16 aP[2], bP[4];   // ping
    BF16x16 aQ[2], bQ[4];   // pong
    loadA(aP, 0);
    loadB(bP, 0);
    // K is a multiple of 64 for all GEMMs in this layer (1024, 4096).
    for (int kb = 0; kb < K; kb += 64) {
        // pull the streaming A lines into cache 8 K-steps ahead
        if (kb + 256 < K) {
            __builtin_prefetch(arow0 + kb + 256, 0, 1);  // global_prefetch_b8
            __builtin_prefetch(arow1 + kb + 256, 0, 1);
        }
        loadA(aQ, kb + 32);
        loadB(bQ, kb + 32);
        mm(aP, bP);
        if (kb + 64 < K) {
            loadA(aP, kb + 64);
            loadB(bP, kb + 64);
        }
        mm(aQ, bQ);
    }

    #pragma unroll
    for (int mt = 0; mt < 2; ++mt) {
        v8f* accs = mt ? acc1 : acc0;
        #pragma unroll
        for (int nt = 0; nt < 4; ++nt) {
            const int col = n0 + nt * 16 + l16;
            const float bvv = bias ? bias[col] : 0.0f;
            #pragma unroll
            for (int r = 0; r < 8; ++r) {
                const int row = m0 + mt * 16 + r + 8 * h;
                float val = accs[nt][r] + bvv;
                if (relu) val = fmaxf(val, 0.0f);
                const size_t o = (size_t)row * N + col;
                if (Cf) Cf[o] = val;
                if (Cb) Cb[o] = f32_to_bf16(val);
            }
        }
    }
}

// ---------------------------------------------------------------------------
// Fused attention with AKT distance decay.
// Grid: (S/16, H, B).  Block: 256 threads (8 waves).
// LDS: score strip 16 x (S+pad) f32  +  16x64 f32 output accumulator.
// ---------------------------------------------------------------------------
#define SROW 2052   // 2048 + 4 pad (bank-conflict stagger)

__global__ void __launch_bounds__(256)
attn_decay_kernel(const unsigned short* __restrict__ Qp,
                  const unsigned short* __restrict__ Kp,
                  const unsigned short* __restrict__ vT,
                  const float* __restrict__ gammas,
                  const int* __restrict__ maskp,
                  unsigned short* __restrict__ concat_b,
                  int Bb, int H, int S, int D) {
    extern __shared__ float smem[];
    float* sc   = smem;                 // 16 * SROW
    float* outb = smem + 16 * SROW;     // 16 * 64

    const int q0 = blockIdx.x * 16;
    const int h  = blockIdx.y;
    const int b  = blockIdx.z;
    const int lane = threadIdx.x & 31;
    const int wv   = threadIdx.x >> 5;
    const int hh   = lane >> 4;
    const int l16  = lane & 15;
    const int maskv = *maskp;
    const float scale = 0.125f;         // 1/sqrt(64)

    // phase 0: zero output accumulator
    for (int i = threadIdx.x; i < 16 * 64; i += 256) outb[i] = 0.0f;

    // ---- phase 1: scores = (Q K^T) * scale, masked, into LDS -------------
    {
        const size_t qrow = ((size_t)b * S + q0 + l16) * D + (size_t)h * 64;
        BF16x16 a0, a1;
        a0.q[0] = *(const uint4*)(Qp + qrow + 8 * hh);
        a0.q[1] = *(const uint4*)(Qp + qrow + 16 + 8 * hh);
        a1.q[0] = *(const uint4*)(Qp + qrow + 32 + 8 * hh);
        a1.q[1] = *(const uint4*)(Qp + qrow + 48 + 8 * hh);

        auto loadK = [&](BF16x16& x0, BF16x16& x1, int k0) {
            const size_t krow =
                ((size_t)b * S + k0 + l16) * D + (size_t)h * 64;
            x0.q[0] = *(const uint4*)(Kp + krow + 8 * hh);
            x0.q[1] = *(const uint4*)(Kp + krow + 8 * hh + 16);
            x1.q[0] = *(const uint4*)(Kp + krow + 32 + 8 * hh);
            x1.q[1] = *(const uint4*)(Kp + krow + 32 + 8 * hh + 16);
        };

        const int NT = S / 16;
        int kt = wv;
        BF16x16 b0c, b1c, b0n, b1n;
        loadK(b0c, b1c, kt * 16);
        while (true) {
            const int ktn = kt + 8;
            if (ktn < NT) {
                loadK(b0n, b1n, ktn * 16);   // prefetch next tile to regs
                if (ktn + 8 < NT) {          // and the one after into cache
                    const size_t pr =
                        ((size_t)b * S + (ktn + 8) * 16 + l16) * D +
                        (size_t)h * 64;
                    __builtin_prefetch(Kp + pr, 0, 1);
                }
            }
            v8f acc = {};
            acc = __builtin_amdgcn_wmma_f32_16x16x32_bf16(
                false, a0.v, false, b0c.v, (short)0, acc, false, false);
            acc = __builtin_amdgcn_wmma_f32_16x16x32_bf16(
                false, a1.v, false, b1c.v, (short)0, acc, false, false);
            const int k0 = kt * 16;
            #pragma unroll
            for (int r = 0; r < 8; ++r) {
                const int m = r + 8 * hh;
                const int i = q0 + m;
                const int j = k0 + l16;
                const bool allow = (j <= i + maskv - 1);
                sc[m * SROW + j] = allow ? acc[r] * scale : NEGF;
            }
            if (ktn >= NT) break;
            b0c = b0n;
            b1c = b1n;
            kt = ktn;
        }
    }
    __syncthreads();

    // ---- phase 2: softmax -> cumsum decay -> re-softmax (per row) --------
    {
        const float g  = gammas[h];
        const float sp = (g > 20.0f) ? g : log1pf(__expf(g));
        const float gamma = -sp;

        for (int rr = wv; rr < 16; rr += 8) {
            float* rowp = sc + rr * SROW;
            const int i = q0 + rr;

            // max
            float m1 = NEGF;
            for (int j = lane; j < S; j += 32) m1 = fmaxf(m1, rowp[j]);
            m1 = wave_max(m1);

            // sums (all entries, and allowed-only for disttotal)
            float s_all = 0.0f, s_msk = 0.0f;
            for (int j = lane; j < S; j += 32) {
                const float e = __expf(rowp[j] - m1);
                s_all += e;
                if (j <= i + maskv - 1) s_msk += e;
            }
            s_all = wave_sum(s_all);
            s_msk = wave_sum(s_msk);
            const float inv = 1.0f / s_all;
            const float disttotal = s_msk * inv;

            // cumsum scan + decay transform + track new max
            float running = 0.0f, m2 = NEGF;
            for (int j0 = 0; j0 < S; j0 += 32) {
                const int j = j0 + lane;
                const float v = rowp[j];
                const bool allow = (j <= i + maskv - 1);
                float p = allow ? __expf(v - m1) * inv : 0.0f;
                float scan = p;
                #pragma unroll
                for (int o = 1; o < 32; o <<= 1) {
                    const float t = __shfl_up(scan, o, 32);
                    if (lane >= o) scan += t;
                }
                const float cum = running + scan;
                running += __shfl(scan, 31, 32);
                const float pe = fabsf((float)(i - j));
                const float ds = sqrtf(fmaxf((disttotal - cum) * pe, 0.0f));
                float eff = __expf(gamma * ds);
                eff = fminf(fmaxf(eff, 1e-5f), 1e5f);
                const float nv = allow ? v * eff : NEGF;
                rowp[j] = nv;
                m2 = fmaxf(m2, nv);
            }
            m2 = wave_max(m2);

            float s2 = 0.0f;
            for (int j = lane; j < S; j += 32) s2 += __expf(rowp[j] - m2);
            s2 = wave_sum(s2);
            const float inv2 = 1.0f / s2;
            for (int j = lane; j < S; j += 32)
                rowp[j] = __expf(rowp[j] - m2) * inv2;
        }
    }
    __syncthreads();

    // ---- phase 3: out = attn @ V via WMMA --------------------------------
    {
        const int nt = wv & 3;                 // 4 N-tiles of dk=64
        const int kstart = (wv >> 2) * 32;     // split K range over wave pairs
        v8f acc = {};
        const size_t vbase =
            (((size_t)b * H + h) * 64 + nt * 16 + l16) * (size_t)S;
        const float* rp = sc + l16 * SROW;     // attn row m = l16

        auto loadV = [&](BF16x16& x, int kb) {
            x.q[0] = *(const uint4*)(vT + vbase + kb + 8 * hh);
            x.q[1] = *(const uint4*)(vT + vbase + kb + 8 * hh + 16);
        };

        BF16x16 bc, bn;
        loadV(bc, kstart);
        for (int kb = kstart; kb < S; kb += 64) {
            const int kbn = kb + 64;
            if (kbn < S) loadV(bn, kbn);       // prefetch next V fragment
            if (kb + 256 < S)                  // and pull stream into cache
                __builtin_prefetch(vT + vbase + kb + 256, 0, 1);
            BF16x16 a;
            #pragma unroll
            for (int t = 0; t < 8; ++t)
                a.u[t] = f32_to_bf16(rp[kb + 8 * hh + t]);
            #pragma unroll
            for (int t = 0; t < 8; ++t)
                a.u[8 + t] = f32_to_bf16(rp[kb + 16 + 8 * hh + t]);
            acc = __builtin_amdgcn_wmma_f32_16x16x32_bf16(
                false, a.v, false, bc.v, (short)0, acc, false, false);
            if (kbn < S) bc = bn;
        }
        #pragma unroll
        for (int r = 0; r < 8; ++r) {
            const int m = r + 8 * hh;
            atomicAdd(&outb[m * 64 + nt * 16 + l16], acc[r]);
        }
    }
    __syncthreads();

    // write 16x64 tile to concat (bf16, (B,S,D) layout, col = h*64 + d)
    for (int e = threadIdx.x; e < 16 * 64; e += 256) {
        const int m = e >> 6, c = e & 63;
        const size_t o = ((size_t)b * S + q0 + m) * D + (size_t)h * 64 + c;
        concat_b[o] = f32_to_bf16(outb[e]);
    }
}

// ---------------------------------------------------------------------------
// Residual add + LayerNorm: out = LN(a + b) * w + bias.
// One block (256 threads) per row, D=1024 (4 elements/thread).
// ---------------------------------------------------------------------------
__global__ void __launch_bounds__(256)
add_ln_kernel(const float* __restrict__ a, const float* __restrict__ bsrc,
              const float* __restrict__ w, const float* __restrict__ bias,
              float* __restrict__ outf, unsigned short* __restrict__ outb,
              int D) {
    const int row = blockIdx.x;
    const int tid = threadIdx.x;
    const int lane = tid & 31, wv = tid >> 5;
    __shared__ float red[8];

    float x[4];
    float s = 0.0f;
    #pragma unroll
    for (int t = 0; t < 4; ++t) {
        const int c = tid + t * 256;
        const float v = a[(size_t)row * D + c] + bsrc[(size_t)row * D + c];
        x[t] = v;
        s += v;
    }
    s = wave_sum(s);
    if (lane == 0) red[wv] = s;
    __syncthreads();
    float tot = 0.0f;
    #pragma unroll
    for (int i = 0; i < 8; ++i) tot += red[i];
    const float mu = tot / (float)D;
    __syncthreads();

    float vs = 0.0f;
    #pragma unroll
    for (int t = 0; t < 4; ++t) {
        const float d = x[t] - mu;
        vs += d * d;
    }
    vs = wave_sum(vs);
    if (lane == 0) red[wv] = vs;
    __syncthreads();
    float vtot = 0.0f;
    #pragma unroll
    for (int i = 0; i < 8; ++i) vtot += red[i];
    const float rstd = rsqrtf(vtot / (float)D + 1e-5f);

    #pragma unroll
    for (int t = 0; t < 4; ++t) {
        const int c = tid + t * 256;
        const float y = (x[t] - mu) * rstd * w[c] + bias[c];
        if (outf) outf[(size_t)row * D + c] = y;
        if (outb) outb[(size_t)row * D + c] = f32_to_bf16(y);
    }
}

// ---------------------------------------------------------------------------
// Host orchestration
// ---------------------------------------------------------------------------
extern "C" void kernel_launch(void* const* d_in, const int* in_sizes, int n_in,
                              void* d_out, int out_size, void* d_ws,
                              size_t ws_size, hipStream_t stream) {
    (void)in_sizes; (void)n_in; (void)out_size; (void)ws_size;
    typedef unsigned short u16;
    const int Bb = 2, S = 2048, D = 1024, H = 16, F = 4096;
    const int M = Bb * S;  // 4096 rows

    const float* query  = (const float*)d_in[0];
    const float* key_   = (const float*)d_in[1];
    const float* values = (const float*)d_in[2];
    const float* Wk = (const float*)d_in[3];
    const float* bk = (const float*)d_in[4];
    const float* Wv = (const float*)d_in[5];
    const float* bv = (const float*)d_in[6];
    const float* Wo = (const float*)d_in[7];
    const float* bo = (const float*)d_in[8];
    const float* gammas = (const float*)d_in[9];
    const float* ln1_w = (const float*)d_in[10];
    const float* ln1_b = (const float*)d_in[11];
    const float* W1 = (const float*)d_in[12];
    const float* b1 = (const float*)d_in[13];
    const float* W2 = (const float*)d_in[14];
    const float* b2 = (const float*)d_in[15];
    const float* ln2_w = (const float*)d_in[16];
    const float* ln2_b = (const float*)d_in[17];
    const int* maskp = (const int*)d_in[18];

    // bump allocator over d_ws (~190 MB total)
    size_t off = 0;
    auto alloc = [&](size_t bytes) -> void* {
        off = (off + 255) & ~(size_t)255;
        void* p = (char*)d_ws + off;
        off += bytes;
        return p;
    };
    u16* qb   = (u16*)alloc((size_t)M * D * 2);
    u16* kb_  = (u16*)alloc((size_t)M * D * 2);
    u16* vb   = (u16*)alloc((size_t)M * D * 2);
    u16* Wk_b = (u16*)alloc((size_t)D * D * 2);
    u16* Wv_b = (u16*)alloc((size_t)D * D * 2);
    u16* Wo_b = (u16*)alloc((size_t)D * D * 2);
    u16* W1_b = (u16*)alloc((size_t)F * D * 2);
    u16* W2_b = (u16*)alloc((size_t)D * F * 2);
    u16* Qp   = (u16*)alloc((size_t)M * D * 2);
    u16* Kp   = (u16*)alloc((size_t)M * D * 2);
    u16* Vp   = (u16*)alloc((size_t)M * D * 2);
    u16* vT   = (u16*)alloc((size_t)M * D * 2);
    u16* concat_b = (u16*)alloc((size_t)M * D * 2);
    float* query2 = (float*)alloc((size_t)M * D * 4);
    float* x_f    = (float*)alloc((size_t)M * D * 4);
    u16*   x_b    = (u16*)alloc((size_t)M * D * 2);
    u16*   h1_b   = (u16*)alloc((size_t)M * F * 2);
    float* ff_f   = (float*)alloc((size_t)M * D * 4);

    const dim3 blk(256);
    auto cvt = [&](const float* in, u16* out, size_t n) {
        cvt_f32_bf16<<<dim3((unsigned)((n + 255) / 256)), blk, 0, stream>>>(
            in, out, n);
    };
    cvt(query,  qb,  (size_t)M * D);
    cvt(key_,   kb_, (size_t)M * D);
    cvt(values, vb,  (size_t)M * D);
    cvt(Wk, Wk_b, (size_t)D * D);
    cvt(Wv, Wv_b, (size_t)D * D);
    cvt(Wo, Wo_b, (size_t)D * D);
    cvt(W1, W1_b, (size_t)F * D);
    cvt(W2, W2_b, (size_t)D * F);

    // projections: Q = query@Wk.T+bk, K = key@Wk.T+bk (kq_same), V = values@Wv.T+bv
    gemm_bf16_wmma<<<dim3(D / 256, M / 64), blk, 0, stream>>>(
        qb, Wk_b, bk, nullptr, Qp, M, D, D, 0);
    gemm_bf16_wmma<<<dim3(D / 256, M / 64), blk, 0, stream>>>(
        kb_, Wk_b, bk, nullptr, Kp, M, D, D, 0);
    gemm_bf16_wmma<<<dim3(D / 256, M / 64), blk, 0, stream>>>(
        vb, Wv_b, bv, nullptr, Vp, M, D, D, 0);

    // V transpose for attn·V B-fragments
    {
        size_t total = (size_t)Bb * D * S;
        transpose_v<<<dim3((unsigned)((total + 255) / 256)), blk, 0, stream>>>(
            Vp, vT, Bb, H, S, D);
    }

    // fused attention + distance decay
    {
        const size_t shmem = (size_t)(16 * SROW + 16 * 64) * sizeof(float);
        attn_decay_kernel<<<dim3(S / 16, H, Bb), blk, shmem, stream>>>(
            Qp, Kp, vT, gammas, maskp, concat_b, Bb, H, S, D);
    }

    // output projection
    gemm_bf16_wmma<<<dim3(D / 256, M / 64), blk, 0, stream>>>(
        concat_b, Wo_b, bo, query2, nullptr, M, D, D, 0);

    // x = LN(query + query2)
    add_ln_kernel<<<dim3(M), blk, 0, stream>>>(
        query, query2, ln1_w, ln1_b, x_f, x_b, D);

    // FFN: h1 = relu(x@W1.T + b1); ff = h1@W2.T + b2
    gemm_bf16_wmma<<<dim3(F / 256, M / 64), blk, 0, stream>>>(
        x_b, W1_b, b1, nullptr, h1_b, M, F, D, 1);
    gemm_bf16_wmma<<<dim3(D / 256, M / 64), blk, 0, stream>>>(
        h1_b, W2_b, b2, ff_f, nullptr, M, D, F, 0);

    // out = LN(x + ff)
    add_ln_kernel<<<dim3(M), blk, 0, stream>>>(
        x_f, ff_f, ln2_w, ln2_b, (float*)d_out, nullptr, D);
}
